// GaussianKernelLayer_52596169507102
// MI455X (gfx1250) — matrix operand order repaired and checked by reading, not compiled, not run on hardware
//
#include <hip/hip_runtime.h>
#include <hip/hip_bf16.h>
#include <math.h>

// Problem constants (match reference)
#define BB  64
#define CC  192
#define KK1 1024
#define KK2 512

typedef __attribute__((ext_vector_type(16))) _Float16 v16h;
typedef __attribute__((ext_vector_type(4)))  float    v4f;
typedef __attribute__((ext_vector_type(8)))  float    v8f;
typedef __attribute__((ext_vector_type(8)))  unsigned v8u;

// ---------------------------------------------------------------------------
// Pre-pass:
//   x2[b,m]  = sum_c x[b,c,m]^2                       (B*K1 floats)
//   c2[n]    = sum_c w[c,n]^2                         (K2 floats)
//   wpk[c,n] = pack( f16(w), f16(w - f16(w)) )        (C*K2 u32, hi|lo split)
// ---------------------------------------------------------------------------
__global__ __launch_bounds__(256) void rbf_pre_kernel(
    const float* __restrict__ x, const float* __restrict__ w,
    float* __restrict__ x2, float* __restrict__ c2,
    unsigned* __restrict__ wpk)
{
    int idx = blockIdx.x * blockDim.x + threadIdx.x;
    if (idx < BB * KK1) {
        int b = idx >> 10;            // / K1
        int m = idx & (KK1 - 1);      // % K1
        const float* p = x + (size_t)b * (CC * KK1) + m;
        float s = 0.0f;
        #pragma unroll 4
        for (int c = 0; c < CC; ++c) {
            float v = p[(size_t)c * KK1];
            s = fmaf(v, v, s);
        }
        x2[idx] = s;
    } else if (idx < BB * KK1 + KK2) {
        int n = idx - BB * KK1;
        float s = 0.0f;
        #pragma unroll 4
        for (int c = 0; c < CC; ++c) {
            float v = w[(size_t)c * KK2 + n];
            s = fmaf(v, v, s);
        }
        c2[n] = s;
    } else if (idx < BB * KK1 + KK2 + CC * KK2) {
        int i = idx - (BB * KK1 + KK2);          // flat (c, n)
        float v = w[i];
        _Float16 h = (_Float16)v;
        _Float16 l = (_Float16)(v - (float)h);
        unsigned hb = (unsigned)__builtin_bit_cast(unsigned short, h);
        unsigned lb = (unsigned)__builtin_bit_cast(unsigned short, l);
        wpk[i] = hb | (lb << 16);
    }
}

// ---------------------------------------------------------------------------
// Main kernel: batched GEMM cross = X^T * W via split-precision f16 WMMA
// (x = xh + xl, w = wh + wl in f16; cross ~= xh*wh + xh*wl + xl*wh, f32 acc,
//  ~2^-22 relative error: fp32-grade), fused Gaussian epilogue, transposed
// (k1-contiguous) 128-bit stores.
//
// Grid: (K1/128, K2/128, B), block 256 = 8 waves.
// Wave tile: 64(M) x 32(N) = 4 x 2 fragments of 16x16, K-step 32.
//
// f16 fragment layouts used here (contiguous k, wave32; lo=lane&15, hi=lane>>4):
//   A 16x32: elem t   <- x[c0 + 16*hi + t    ][M=lo]   (t=0..7)
//            elem 8+t <- x[c0 + 16*hi + 8 + t][M=lo]
//   B 32x16: elem e   <- w[c0 + 16*hi + e    ][N=lo]   (e=0..15)
//   C/D 16x16 f32: vgpr r, lane l -> M = r + 8*hi, N = lo
// ---------------------------------------------------------------------------
__global__ __launch_bounds__(256) void rbf_main_kernel(
    const float* __restrict__ x, const unsigned* __restrict__ wpk,
    const float* __restrict__ sigma,
    const float* __restrict__ x2, const float* __restrict__ c2,
    float* __restrict__ out)
{
    const int lane = threadIdx.x & 31;
    const int wv   = threadIdx.x >> 5;
    const int lo   = lane & 15;
    const int hi   = lane >> 4;
    const int waveM = wv & 1;          // 0..1
    const int waveN = wv >> 1;         // 0..3
    const int b = blockIdx.z;

    const int mBase = blockIdx.x * 128 + waveM * 64;   // wave M origin
    const int nBase = blockIdx.y * 128 + waveN * 32;   // wave N origin

    const float* xb = x + (size_t)b * (CC * KK1);

    v8f acc[4][2] = {};   // [mi][ni]

    for (int c0 = 0; c0 < CC; c0 += 32) {
        // ---- B fragments from pre-split packed hi|lo u32 plane ----
        v16h Bh[2], Bl[2];
        {
            const unsigned* wp = wpk + (size_t)(c0 + 16 * hi) * KK2 + (nBase + lo);
            #pragma unroll
            for (int ni = 0; ni < 2; ++ni) {
                v8u bh, bl;
                #pragma unroll
                for (int j = 0; j < 8; ++j) {
                    unsigned r0 = wp[(size_t)(2 * j)     * KK2 + ni * 16]; // c=c0+16hi+2j
                    unsigned r1 = wp[(size_t)(2 * j + 1) * KK2 + ni * 16];
                    bh[j] = __builtin_amdgcn_perm(r1, r0, 0x05040100u); // {r1.lo16, r0.lo16}
                    bl[j] = __builtin_amdgcn_perm(r1, r0, 0x07060302u); // {r1.hi16, r0.hi16}
                }
                Bh[ni] = __builtin_bit_cast(v16h, bh);
                Bl[ni] = __builtin_bit_cast(v16h, bl);
            }
        }

        #pragma unroll
        for (int mi = 0; mi < 4; ++mi) {
            // ---- A fragment (hi/lo split, inline conversion) ----
            v16h Ah, Al;
            const float* xp = xb + (size_t)(c0 + 16 * hi) * KK1
                                 + (mBase + mi * 16 + lo);
            #pragma unroll
            for (int t = 0; t < 8; ++t) {
                float v0 = xp[(size_t)t * KK1];          // c = c0+16hi+t
                float v1 = xp[(size_t)(t + 8) * KK1];    // c = c0+16hi+8+t
                _Float16 h0 = (_Float16)v0;
                _Float16 h1 = (_Float16)v1;
                Ah[t]     = h0;
                Al[t]     = (_Float16)(v0 - (float)h0);
                Ah[8 + t] = h1;
                Al[8 + t] = (_Float16)(v1 - (float)h1);
            }

            // ---- 3-product split-precision accumulate ----
            #pragma unroll
            for (int ni = 0; ni < 2; ++ni) {
                acc[mi][ni] = __builtin_amdgcn_wmma_f32_16x16x32_f16(
                    false, Ah, false, Bh[ni], (short)0, acc[mi][ni], false, false);
                acc[mi][ni] = __builtin_amdgcn_wmma_f32_16x16x32_f16(
                    false, Ah, false, Bl[ni], (short)0, acc[mi][ni], false, false);
                acc[mi][ni] = __builtin_amdgcn_wmma_f32_16x16x32_f16(
                    false, Al, false, Bh[ni], (short)0, acc[mi][ni], false, false);
            }
        }
    }

    const float s  = sigma[0];
    const float sc = -0.5f / (s * s);

    #pragma unroll
    for (int mi = 0; mi < 4; ++mi) {
        const int mrow = mBase + mi * 16 + 8 * hi;      // 8 consecutive rows
        const float* x2p = x2 + (size_t)b * KK1 + mrow;
        v4f xq0 = *(const v4f*)(x2p);
        v4f xq1 = *(const v4f*)(x2p + 4);
        #pragma unroll
        for (int ni = 0; ni < 2; ++ni) {
            const int ncol = nBase + ni * 16 + lo;
            const float c2v = c2[ncol];
            float* op = out + ((size_t)b * KK2 + ncol) * KK1 + mrow;
            v4f o0, o1;
            #pragma unroll
            for (int r = 0; r < 4; ++r) {
                float sq = fmaxf(xq0[r] + c2v - 2.0f * acc[mi][ni][r], 0.0f);
                o0[r] = __expf(sc * sq);
            }
            #pragma unroll
            for (int r = 0; r < 4; ++r) {
                float sq = fmaxf(xq1[r] + c2v - 2.0f * acc[mi][ni][4 + r], 0.0f);
                o1[r] = __expf(sc * sq);
            }
            *(v4f*)(op)     = o0;
            *(v4f*)(op + 4) = o1;
        }
    }
}

// ---------------------------------------------------------------------------
extern "C" void kernel_launch(void* const* d_in, const int* in_sizes, int n_in,
                              void* d_out, int out_size, void* d_ws, size_t ws_size,
                              hipStream_t stream) {
    const float* x     = (const float*)d_in[0];   // (B, C, K1)
    const float* w3c   = (const float*)d_in[1];   // (1, C, K2)
    const float* sigma = (const float*)d_in[2];   // (1,)
    float* out = (float*)d_out;                   // (B, K2, K1)

    // Workspace layout (~660 KB total):
    float*    x2  = (float*)d_ws;                 // B*K1 floats
    float*    c2  = x2 + BB * KK1;                // K2 floats
    unsigned* wpk = (unsigned*)(c2 + KK2);        // C*K2 u32 (packed hi|lo f16)

    {
        int total = BB * KK1 + KK2 + CC * KK2;
        int blocks = (total + 255) / 256;
        rbf_pre_kernel<<<blocks, 256, 0, stream>>>(x, w3c, x2, c2, wpk);
    }
    {
        dim3 grid(KK1 / 128, KK2 / 128, BB);
        rbf_main_kernel<<<grid, 256, 0, stream>>>(x, wpk, sigma, x2, c2, out);
    }
}
